// HEPTOCModel_22067541966908
// MI455X (gfx1250) — matrix operand order repaired and checked by reading, not compiled, not run on hardware
//
#include <hip/hip_runtime.h>

typedef __attribute__((ext_vector_type(16))) _Float16 v16h;
typedef __attribute__((ext_vector_type(8)))  _Float16 v8h;
typedef __attribute__((ext_vector_type(8)))  float    v8f;

#define NLAYER 4
#define NHEAD  8
#define DHEAD  64
#define HDIM   512
#define BLK    128
#define NROUND 3
#define SSCALE 0.125f   // 1/sqrt(64)

// ---------------------------------------------------------------------------
// WMMA helpers (gfx1250 layouts per CDNA5 ISA 7.12.2)
// ---------------------------------------------------------------------------
__device__ __forceinline__ v8f wmma32(v16h a, v16h b, v8f c) {
  return __builtin_amdgcn_wmma_f32_16x16x32_f16(false, a, false, b, (short)0, c,
                                                false, false);
}
__device__ __forceinline__ v8f zero8() {
  v8f z = {0.f, 0.f, 0.f, 0.f, 0.f, 0.f, 0.f, 0.f};
  return z;
}
// A fragment 16x32 f16: p points at row (lane&15), col k0 + (lane>>4)*16
// elements 0..15 = K k0+half*16 .. +15 (contiguous)
__device__ __forceinline__ v16h frag_a(const _Float16* p) {
  v16h r;
  ((v8h*)&r)[0] = *(const v8h*)(p);
  ((v8h*)&r)[1] = *(const v8h*)(p + 8);
  return r;
}
// B fragment 32x16 f16 from transposed (NxK) storage:
// p points at row n=(lane&15), col k0 + (lane>>4)*16 ; 16 contiguous K values
__device__ __forceinline__ v16h frag_b(const _Float16* p) {
  v16h r;
  ((v8h*)&r)[0] = *(const v8h*)(p);
  ((v8h*)&r)[1] = *(const v8h*)(p + 8);
  return r;
}

// ---------------------------------------------------------------------------
// Generic WMMA GEMM: out = act(A[MxK] @ B[KxN] + bias) (+res)
// A: f16 row-major lda=K.  Bt: f16 transposed (N x K) row-major.
// Wave computes a 16 x (NT*16) tile; NT is compile-time so the WMMA loop is
// branch-free (no exec masking, accumulators stay in fixed VGPRs).
// ---------------------------------------------------------------------------
template <int NT>
__global__ __launch_bounds__(256) void gemm_f16_kernel(
    const _Float16* __restrict__ A, const _Float16* __restrict__ Bt,
    const float* __restrict__ bias, const float* __restrict__ res,
    float* __restrict__ outF, _Float16* __restrict__ outH,
    int M, int N, int K, int ldF, int ldH, int relu)
{
  const int lane = threadIdx.x & 31;
  const int wid  = threadIdx.x >> 5;
  const int cg   = N / (NT * 16);          // col groups (uniform per launch)
  const long wg  = (long)blockIdx.x * 8 + wid;
  const int mtile = (int)(wg / cg);
  if (mtile * 16 >= M) return;
  const int col0 = (int)(wg % cg) * (NT * 16);
  const int row0 = mtile * 16;
  const int m = lane & 15, half = lane >> 4;

  v8f acc[NT];
#pragma unroll
  for (int i = 0; i < NT; ++i) acc[i] = zero8();

  const _Float16* arow = A + (size_t)(row0 + m) * K + half * 16;
  const _Float16* brow = Bt + (size_t)(col0 + m) * K + half * 16;
  for (int k0 = 0; k0 < K; k0 += 32) {
    v16h af = frag_a(arow + k0);
#pragma unroll
    for (int nt = 0; nt < NT; ++nt)
      acc[nt] = wmma32(af, frag_b(brow + (size_t)nt * 16 * K + k0), acc[nt]);
  }
#pragma unroll
  for (int nt = 0; nt < NT; ++nt) {
    const int col = col0 + nt * 16 + m;
    const float bv = bias ? bias[col] : 0.f;
#pragma unroll
    for (int r = 0; r < 8; ++r) {
      const int row = row0 + r + half * 8;   // C layout: M = r + 8*(lane>>4)
      float v = acc[nt][r] + bv;
      if (res)  v += res[(size_t)row * N + col];
      if (relu) v = fmaxf(v, 0.f);
      if (outF) outF[(size_t)row * ldF + col] = v;
      if (outH) outH[(size_t)row * ldH + col] = (_Float16)v;
    }
  }
}

// ---------------------------------------------------------------------------
// Fused HEPT block attention for one (block b, head h), one hash round.
// Reads q/k/v in *unsorted* order via idx[]; scatters o/lse back unsorted.
// ---------------------------------------------------------------------------
__global__ __launch_bounds__(256) void attn_kernel(
    const _Float16* __restrict__ q, const _Float16* __restrict__ k,
    const _Float16* __restrict__ v, const int* __restrict__ idx,
    const float* __restrict__ coords, const int* __restrict__ batch,
    const float* __restrict__ whead, _Float16* __restrict__ o_out,
    float* __restrict__ lse_out)
{
  __shared__ _Float16 vS[64 * 136];        // v^T tile  [d][t], stride 136
  __shared__ _Float16 pS[8 * 16 * 136];    // per-wave p tiles, stride 136
  __shared__ float c1S[BLK];
  __shared__ int   bS[BLK];
  __shared__ int   idxS[BLK];

  const int b = blockIdx.x, h = blockIdx.y;
  const int tid = threadIdx.x;

  if (tid < BLK) {
    int tk = idx[b * BLK + tid];
    idxS[tid] = tk;
    c1S[tid]  = coords[2 * tk + 1];
    bS[tid]   = batch[tk];
  }
  __syncthreads();

  // transpose v block into LDS: vS[d*136 + t]
  for (int it = tid; it < 1024; it += 256) {
    const int t = it >> 3, seg = it & 7;
    const v8h vv = *(const v8h*)(v + (size_t)idxS[t] * HDIM + h * DHEAD + seg * 8);
#pragma unroll
    for (int j = 0; j < 8; ++j) vS[(seg * 8 + j) * 136 + t] = vv[j];
  }

  const int w = tid >> 5, lane = tid & 31, m = lane & 15, half = lane >> 4;
  const int ibase = w * 16;
  const float wh = whead[h];

  // q fragments for this wave's 16-row strip (K = 64 -> two frags)
  const _Float16* qp = q + (size_t)idxS[ibase + m] * HDIM + h * DHEAD + half * 16;
  v16h aq0 = frag_a(qp);
  v16h aq1 = frag_a(qp + 32);

  float c_i[8]; int b_i[8];
#pragma unroll
  for (int r = 0; r < 8; ++r) {
    const int row = ibase + r + half * 8;
    c_i[r] = c1S[row];  b_i[r] = bS[row];
  }

  // scores: 8 j-tiles of 16 columns
  float s[8][8];
#pragma unroll
  for (int jt = 0; jt < 8; ++jt) {
    const int j = jt * 16 + m;
    const _Float16* kp = k + (size_t)idxS[j] * HDIM + h * DHEAD + half * 16;
    v8f acc = zero8();
    acc = wmma32(aq0, frag_b(kp), acc);
    acc = wmma32(aq1, frag_b(kp + 32), acc);
    const float cj = c1S[j];  const int bj = bS[j];
#pragma unroll
    for (int r = 0; r < 8; ++r) {
      const float d = c_i[r] - cj;
      const float val = acc[r] * SSCALE - wh * d * d;
      s[jt][r] = (b_i[r] == bj) ? val : -1.0e9f;
    }
  }

  // row-wise softmax stats (row M spans a 16-lane half-wave)
  float rmax[8], rsum[8];
#pragma unroll
  for (int r = 0; r < 8; ++r) {
    float mx = s[0][r];
#pragma unroll
    for (int jt = 1; jt < 8; ++jt) mx = fmaxf(mx, s[jt][r]);
    mx = fmaxf(mx, __shfl_xor(mx, 1, 16));
    mx = fmaxf(mx, __shfl_xor(mx, 2, 16));
    mx = fmaxf(mx, __shfl_xor(mx, 4, 16));
    mx = fmaxf(mx, __shfl_xor(mx, 8, 16));
    float sm = 0.f;
#pragma unroll
    for (int jt = 0; jt < 8; ++jt) sm += __expf(s[jt][r] - mx);
    sm += __shfl_xor(sm, 1, 16);
    sm += __shfl_xor(sm, 2, 16);
    sm += __shfl_xor(sm, 4, 16);
    sm += __shfl_xor(sm, 8, 16);
    rmax[r] = mx;  rsum[r] = sm;
  }
  if (m == 0) {
#pragma unroll
    for (int r = 0; r < 8; ++r) {
      const int row = ibase + r + half * 8;
      lse_out[(size_t)idxS[row] * NHEAD + h] = rmax[r] + __logf(rsum[r]);
    }
  }

  // p = exp(s - rowmax) -> LDS (A-layout round trip)
  _Float16* pw = pS + w * (16 * 136);
#pragma unroll
  for (int jt = 0; jt < 8; ++jt)
#pragma unroll
    for (int r = 0; r < 8; ++r)
      pw[(r + half * 8) * 136 + jt * 16 + m] = (_Float16)__expf(s[jt][r] - rmax[r]);
  __syncthreads();   // covers vS fill + own p tile

  // o = p @ v  (K = 128 -> 4 chunks; N = 64 -> 4 d-tiles)
  v8f oacc[4];
#pragma unroll
  for (int i = 0; i < 4; ++i) oacc[i] = zero8();
#pragma unroll
  for (int kc = 0; kc < 4; ++kc) {
    v16h ap = frag_a(pw + m * 136 + kc * 32 + half * 16);
#pragma unroll
    for (int dt = 0; dt < 4; ++dt)
      oacc[dt] = wmma32(ap, frag_b(vS + (dt * 16 + m) * 136 + kc * 32 + half * 16),
                        oacc[dt]);
  }
#pragma unroll
  for (int r = 0; r < 8; ++r) {
    const int row = ibase + r + half * 8;
    const size_t ob = (size_t)idxS[row] * HDIM + h * DHEAD;
    const float inv = 1.f / rsum[r];
#pragma unroll
    for (int dt = 0; dt < 4; ++dt)
      o_out[ob + dt * 16 + m] = (_Float16)(oacc[dt][r] * inv);
  }
}

// ---------------------------------------------------------------------------
// Combine 3 hash rounds with softmax(lse) weights
// ---------------------------------------------------------------------------
__global__ __launch_bounds__(256) void combine_kernel(
    const _Float16* __restrict__ o3, const float* __restrict__ lse3,
    _Float16* __restrict__ att, int N)
{
  const size_t i = (size_t)blockIdx.x * 256 + threadIdx.x;
  const int n = (int)(i >> 9), e = (int)(i & 511), h = e >> 6;
  const float l0 = lse3[(size_t)n * 8 + h];
  const float l1 = lse3[(size_t)N * 8 + (size_t)n * 8 + h];
  const float l2 = lse3[2 * (size_t)N * 8 + (size_t)n * 8 + h];
  const float mx = fmaxf(l0, fmaxf(l1, l2));
  const float w0 = __expf(l0 - mx), w1 = __expf(l1 - mx), w2 = __expf(l2 - mx);
  const float is = 1.f / (w0 + w1 + w2);
  const size_t off = (size_t)n * HDIM + e;
  const float o = w0 * (float)o3[off] + w1 * (float)o3[(size_t)N * HDIM + off] +
                  w2 * (float)o3[2 * (size_t)N * HDIM + off];
  att[off] = (_Float16)(o * is);
}

// ---------------------------------------------------------------------------
// LayerNorm over 64 dims, one wave per token, f16 output
// ---------------------------------------------------------------------------
__global__ __launch_bounds__(256) void ln_kernel(
    const float* __restrict__ x, const float* __restrict__ g,
    const float* __restrict__ bb, _Float16* __restrict__ out)
{
  const int w = threadIdx.x >> 5, lane = threadIdx.x & 31;
  const int tok = blockIdx.x * 8 + w;
  const float* xr = x + (size_t)tok * 64;
  const float a0 = xr[lane], a1 = xr[lane + 32];
  float sm = a0 + a1;
  for (int off = 16; off >= 1; off >>= 1) sm += __shfl_xor(sm, off, 32);
  const float mean = sm * (1.f / 64.f);
  const float d0 = a0 - mean, d1 = a1 - mean;
  float vv = d0 * d0 + d1 * d1;
  for (int off = 16; off >= 1; off >>= 1) vv += __shfl_xor(vv, off, 32);
  const float rs = rsqrtf(vv * (1.f / 64.f) + 1e-5f);
  out[(size_t)tok * 64 + lane]      = (_Float16)(d0 * rs * g[lane] + bb[lane]);
  out[(size_t)tok * 64 + lane + 32] = (_Float16)(d1 * rs * g[lane + 32] + bb[lane + 32]);
}

// ---------------------------------------------------------------------------
// Weight prep: f32 (K x N) -> f16 transposed (N x Kpad), zero-padded K
// ---------------------------------------------------------------------------
__global__ void transpose_f16_kernel(const float* __restrict__ W,
                                     _Float16* __restrict__ Wt,
                                     int K, int Kpad, int Ncols)
{
  const int i = blockIdx.x * 256 + threadIdx.x;
  if (i >= Ncols * Kpad) return;
  const int n = i / Kpad, kk = i % Kpad;
  Wt[i] = (kk < K) ? (_Float16)W[(size_t)kk * Ncols + n] : (_Float16)0.f;
}

__global__ void pad_x_kernel(const float* __restrict__ x, _Float16* __restrict__ xp, int N)
{
  const int i = blockIdx.x * 256 + threadIdx.x;
  if (i >= N * 32) return;
  const int n = i >> 5, c = i & 31;
  xp[i] = (c < 16) ? (_Float16)x[(size_t)n * 16 + c] : (_Float16)0.f;
}

__global__ void whead_kernel(const float* __restrict__ rpe, float* __restrict__ wh)
{
  const int t = threadIdx.x;            // 32 = L*H
  if (t >= 32) return;
  const int l = t >> 3, h = t & 7;
  const float* p = rpe + (size_t)l * 8 * HDIM + h * DHEAD;
  float s = 0.f;
  for (int w2 = 0; w2 < 8; ++w2)
    for (int d = 0; d < DHEAD; ++d) s += p[w2 * HDIM + d];
  s *= (1.f / 512.f);
  wh[t] = (s > 20.f) ? s : log1pf(__expf(s));   // softplus
}

// ---------------------------------------------------------------------------
// Sort-key preparation + bitonic argsort (stable via packed index)
// ---------------------------------------------------------------------------
__global__ __launch_bounds__(256) void minmax_part_kernel(
    const float* __restrict__ coords, const float* __restrict__ hp,
    float* __restrict__ part, int N)
{
  __shared__ float sh[8][256];
  float mn0 = 3e38f, mn1 = 3e38f, mn2 = 3e38f, mn3 = 3e38f;
  float mx0 = -3e38f, mx1 = -3e38f, mx2 = -3e38f, mx3 = -3e38f;
  const float h0x = hp[0], h0y = hp[1], h1x = hp[2], h1y = hp[3], h2x = hp[4], h2y = hp[5];
  for (int n = blockIdx.x * 256 + threadIdx.x; n < N; n += gridDim.x * 256) {
    const float c0 = coords[2 * n], c1 = coords[2 * n + 1];
    const float v1 = c0 * h0x + c1 * h0y;
    const float v2 = c0 * h1x + c1 * h1y;
    const float v3 = c0 * h2x + c1 * h2y;
    mn0 = fminf(mn0, c0); mx0 = fmaxf(mx0, c0);
    mn1 = fminf(mn1, v1); mx1 = fmaxf(mx1, v1);
    mn2 = fminf(mn2, v2); mx2 = fmaxf(mx2, v2);
    mn3 = fminf(mn3, v3); mx3 = fmaxf(mx3, v3);
  }
  const int t = threadIdx.x;
  sh[0][t] = mn0; sh[1][t] = mn1; sh[2][t] = mn2; sh[3][t] = mn3;
  sh[4][t] = mx0; sh[5][t] = mx1; sh[6][t] = mx2; sh[7][t] = mx3;
  __syncthreads();
  for (int s2 = 128; s2 > 0; s2 >>= 1) {
    if (t < s2) {
      for (int j = 0; j < 4; ++j) sh[j][t] = fminf(sh[j][t], sh[j][t + s2]);
      for (int j = 4; j < 8; ++j) sh[j][t] = fmaxf(sh[j][t], sh[j][t + s2]);
    }
    __syncthreads();
  }
  if (t < 8) part[blockIdx.x * 8 + t] = sh[t][0];
}

__global__ void minmax_final_kernel(const float* __restrict__ part,
                                    float* __restrict__ fin, int nb)
{
  const int t = threadIdx.x;
  if (t < 4) {
    float m = 3e38f;
    for (int b2 = 0; b2 < nb; ++b2) m = fminf(m, part[b2 * 8 + t]);
    fin[t] = m;
  } else if (t < 8) {
    float m = -3e38f;
    for (int b2 = 0; b2 < nb; ++b2) m = fmaxf(m, part[b2 * 8 + t]);
    fin[t] = m;
  }
}

__global__ void make_keys_kernel(const float* __restrict__ coords,
                                 const int* __restrict__ batch,
                                 const float* __restrict__ hp,
                                 const float* __restrict__ fin,
                                 unsigned long long* __restrict__ keys, int N)
{
  const int n = blockIdx.x * 256 + threadIdx.x;
  if (n >= N) return;
  const float c0 = coords[2 * n], c1 = coords[2 * n + 1];
  int region = (int)((c0 - fin[0]) / (fin[4] - fin[0] + 1e-6f) * 8.0f);
  region = min(max(region, 0), 7);
  const float base = ((float)batch[n] * 8.0f + (float)region) * 2.0f;
  for (int r = 0; r < NROUND; ++r) {
    const float hv = c0 * hp[2 * r] + c1 * hp[2 * r + 1];
    const float hn = (hv - fin[1 + r]) / (fin[5 + r] - fin[1 + r] + 1e-6f);
    const unsigned int bits = __float_as_uint(base + hn);   // skey >= 0: monotonic
    keys[(size_t)r * N + n] = ((unsigned long long)bits << 32) | (unsigned int)n;
  }
}

__global__ void bitonic_pass_kernel(unsigned long long* __restrict__ keys,
                                    int N, int j, int kk)
{
  const int i = blockIdx.x * 256 + threadIdx.x;
  unsigned long long* a = keys + (size_t)blockIdx.y * N;
  const int ixj = i ^ j;
  if (ixj > i) {
    const unsigned long long x = a[i], y = a[ixj];
    const bool asc = ((i & kk) == 0);
    if ((x > y) == asc) { a[i] = y; a[ixj] = x; }
  }
}

__global__ void extract_idx_kernel(const unsigned long long* __restrict__ keys,
                                   int* __restrict__ idx, int N)
{
  const int pos = blockIdx.x * 256 + threadIdx.x;
  const int r = blockIdx.y;
  if (pos < N)
    idx[(size_t)r * N + pos] = (int)(keys[(size_t)r * N + pos] & 0xffffffffull);
}

// ---------------------------------------------------------------------------
// Final tiny heads: beta = clip(sigmoid(t2 @ bW2 + bb2)), hcoord = t3 @ cW2 + cb2
// ---------------------------------------------------------------------------
__global__ void heads_kernel(const _Float16* __restrict__ t2,
                             const _Float16* __restrict__ t3,
                             const float* __restrict__ bW2, const float* __restrict__ bb2,
                             const float* __restrict__ cW2, const float* __restrict__ cb2,
                             float* __restrict__ outp, int N)
{
  const int n = blockIdx.x * 256 + threadIdx.x;
  if (n >= N) return;
  float sb = bb2[0];
  for (int i = 0; i < 16; ++i) sb += (float)t2[(size_t)n * 16 + i] * bW2[i];
  float beta = 1.f / (1.f + __expf(-sb));
  beta = fminf(fmaxf(beta, 1e-6f), 1.f - 1e-6f);
  outp[n] = beta;
  for (int j = 0; j < 3; ++j) {
    float s = cb2[j];
    for (int i = 0; i < 16; ++i) s += (float)t3[(size_t)n * 16 + i] * cW2[i * 3 + j];
    outp[(size_t)N + (size_t)n * 3 + j] = s;
  }
}

// ---------------------------------------------------------------------------
// Host orchestration
// ---------------------------------------------------------------------------
extern "C" void kernel_launch(void* const* d_in, const int* in_sizes, int n_in,
                              void* d_out, int out_size, void* d_ws, size_t ws_size,
                              hipStream_t stream) {
  const float* x        = (const float*)d_in[0];
  const float* coords   = (const float*)d_in[1];
  const int*   batch    = (const int*)  d_in[2];
  const float* enc_W1   = (const float*)d_in[3];
  const float* enc_b1   = (const float*)d_in[4];
  const float* enc_W2   = (const float*)d_in[5];
  const float* enc_b2   = (const float*)d_in[6];
  const float* ln1_g    = (const float*)d_in[7];
  const float* ln1_b    = (const float*)d_in[8];
  const float* wq       = (const float*)d_in[9];
  const float* wk       = (const float*)d_in[10];
  const float* wv       = (const float*)d_in[11];
  const float* rpe_W    = (const float*)d_in[12];
  const float* wo       = (const float*)d_in[13];
  const float* ln2_g    = (const float*)d_in[14];
  const float* ln2_b    = (const float*)d_in[15];
  const float* ff_W1    = (const float*)d_in[16];
  const float* ff_b1    = (const float*)d_in[17];
  const float* ff_W2    = (const float*)d_in[18];
  const float* ff_b2    = (const float*)d_in[19];
  const float* W_cat    = (const float*)d_in[20];
  const float* mlp_W1   = (const float*)d_in[21];
  const float* mlp_b1   = (const float*)d_in[22];
  const float* mlp_W2   = (const float*)d_in[23];
  const float* mlp_b2   = (const float*)d_in[24];
  const float* beta_W1  = (const float*)d_in[25];
  const float* beta_b1  = (const float*)d_in[26];
  const float* beta_W2  = (const float*)d_in[27];
  const float* beta_b2  = (const float*)d_in[28];
  const float* coord_W1 = (const float*)d_in[29];
  const float* coord_b1 = (const float*)d_in[30];
  const float* coord_W2 = (const float*)d_in[31];
  const float* coord_b2 = (const float*)d_in[32];
  const float* hash_pj  = (const float*)d_in[33];

  const int N = in_sizes[0] / 16;   // 32768 tokens

  // ---- bump allocator over workspace ----
  char* base = (char*)d_ws;
  size_t off = 0;
  auto alloc = [&](size_t bytes) -> void* {
    void* p = base + off;
    off += (bytes + 255) & ~(size_t)255;
    return p;
  };
  _Float16* wq_t  = (_Float16*)alloc((size_t)NLAYER * HDIM * DHEAD * 2);
  _Float16* wk_t  = (_Float16*)alloc((size_t)NLAYER * HDIM * DHEAD * 2);
  _Float16* wv_t  = (_Float16*)alloc((size_t)NLAYER * HDIM * DHEAD * 2);
  _Float16* wo_t  = (_Float16*)alloc((size_t)NLAYER * DHEAD * HDIM * 2);
  _Float16* ff1_t = (_Float16*)alloc((size_t)NLAYER * 64 * 64 * 2);
  _Float16* ff2_t = (_Float16*)alloc((size_t)NLAYER * 64 * 64 * 2);
  _Float16* eW1_t = (_Float16*)alloc((size_t)64 * 32 * 2);
  _Float16* eW2_t = (_Float16*)alloc((size_t)64 * 64 * 2);
  _Float16* cat_t = (_Float16*)alloc((size_t)32 * 320 * 2);
  _Float16* m1_t  = (_Float16*)alloc((size_t)32 * 32 * 2);
  _Float16* m2_t  = (_Float16*)alloc((size_t)32 * 32 * 2);
  _Float16* bW1_t = (_Float16*)alloc((size_t)16 * 32 * 2);
  _Float16* cW1_t = (_Float16*)alloc((size_t)16 * 32 * 2);
  float*    whead = (float*)alloc(32 * 4);
  float*    parts = (float*)alloc(64 * 8 * 4);
  float*    fins  = (float*)alloc(8 * 4);
  unsigned long long* keys = (unsigned long long*)alloc((size_t)NROUND * N * 8);
  int*      idx   = (int*)alloc((size_t)NROUND * N * 4);
  _Float16* xpad  = (_Float16*)alloc((size_t)N * 32 * 2);
  _Float16* henc  = (_Float16*)alloc((size_t)N * 64 * 2);
  float*    cur   = (float*)alloc((size_t)N * 64 * 4);
  _Float16* xn    = (_Float16*)alloc((size_t)N * 64 * 2);
  _Float16* qb    = (_Float16*)alloc((size_t)N * HDIM * 2);
  _Float16* kb    = (_Float16*)alloc((size_t)N * HDIM * 2);
  _Float16* vb    = (_Float16*)alloc((size_t)N * HDIM * 2);
  _Float16* o3    = (_Float16*)alloc((size_t)NROUND * N * HDIM * 2);
  float*    lse3  = (float*)alloc((size_t)NROUND * N * NHEAD * 4);
  _Float16* att   = (_Float16*)alloc((size_t)N * HDIM * 2);
  _Float16* ffh   = (_Float16*)alloc((size_t)N * 64 * 2);
  _Float16* cat   = (_Float16*)alloc((size_t)N * 320 * 2);
  float*    e2F   = (float*)alloc((size_t)N * 32 * 4);
  _Float16* e2H   = (_Float16*)alloc((size_t)N * 32 * 2);
  float*    moF   = (float*)alloc((size_t)N * 32 * 4);
  _Float16* moH   = (_Float16*)alloc((size_t)N * 32 * 2);
  _Float16* t1    = (_Float16*)alloc((size_t)N * 32 * 2);
  _Float16* t2b   = (_Float16*)alloc((size_t)N * 16 * 2);
  _Float16* t3b   = (_Float16*)alloc((size_t)N * 16 * 2);
  (void)ws_size; (void)n_in; (void)out_size;

  // NT (16-col tiles per wave) is compile-time; dispatch on N_
  auto gemm = [&](const _Float16* A, const _Float16* Bt, const float* bias,
                  const float* res, float* outF, int ldF, _Float16* outH, int ldH,
                  int M_, int N_, int K_, int relu) {
    const int nt = (N_ >= 64) ? 4 : (N_ >> 4);
    const int cg = N_ / (nt * 16);
    const long waves = (long)(M_ / 16) * cg;
    const int blocks = (int)((waves + 7) / 8);
    if (nt == 4)
      gemm_f16_kernel<4><<<blocks, 256, 0, stream>>>(A, Bt, bias, res, outF, outH,
                                                     M_, N_, K_, ldF, ldH, relu);
    else if (nt == 2)
      gemm_f16_kernel<2><<<blocks, 256, 0, stream>>>(A, Bt, bias, res, outF, outH,
                                                     M_, N_, K_, ldF, ldH, relu);
    else
      gemm_f16_kernel<1><<<blocks, 256, 0, stream>>>(A, Bt, bias, res, outF, outH,
                                                     M_, N_, K_, ldF, ldH, relu);
  };
  auto tconv = [&](const float* W, _Float16* Wt, int K, int Kpad, int Nc) {
    const int tot = Nc * Kpad;
    transpose_f16_kernel<<<(tot + 255) / 256, 256, 0, stream>>>(W, Wt, K, Kpad, Nc);
  };

  // ---- weight prep (deterministic each call) ----
  for (int l = 0; l < NLAYER; ++l) {
    tconv(wq + (size_t)l * 64 * 512, wq_t + (size_t)l * 512 * 64, 64, 64, 512);
    tconv(wk + (size_t)l * 64 * 512, wk_t + (size_t)l * 512 * 64, 64, 64, 512);
    tconv(wv + (size_t)l * 64 * 512, wv_t + (size_t)l * 512 * 64, 64, 64, 512);
    tconv(wo + (size_t)l * 512 * 64, wo_t + (size_t)l * 64 * 512, 512, 512, 64);
    tconv(ff_W1 + (size_t)l * 64 * 64, ff1_t + (size_t)l * 64 * 64, 64, 64, 64);
    tconv(ff_W2 + (size_t)l * 64 * 64, ff2_t + (size_t)l * 64 * 64, 64, 64, 64);
  }
  tconv(enc_W1, eW1_t, 16, 32, 64);
  tconv(enc_W2, eW2_t, 64, 64, 64);
  tconv(W_cat, cat_t, 320, 320, 32);
  tconv(mlp_W1, m1_t, 32, 32, 32);
  tconv(mlp_W2, m2_t, 32, 32, 32);
  tconv(beta_W1, bW1_t, 32, 32, 16);
  tconv(coord_W1, cW1_t, 32, 32, 16);
  whead_kernel<<<1, 32, 0, stream>>>(rpe_W, whead);

  // ---- sort keys + bitonic argsort (3 rounds batched on grid.y) ----
  minmax_part_kernel<<<64, 256, 0, stream>>>(coords, hash_pj, parts, N);
  minmax_final_kernel<<<1, 32, 0, stream>>>(parts, fins, 64);
  make_keys_kernel<<<(N + 255) / 256, 256, 0, stream>>>(coords, batch, hash_pj,
                                                        fins, keys, N);
  for (int kk = 2; kk <= N; kk <<= 1)
    for (int j = kk >> 1; j > 0; j >>= 1)
      bitonic_pass_kernel<<<dim3(N / 256, NROUND), 256, 0, stream>>>(keys, N, j, kk);
  extract_idx_kernel<<<dim3(N / 256, NROUND), 256, 0, stream>>>(keys, idx, N);

  // ---- encoder: cur = h0 = relu(x@W1+b1)@W2+b2 ; feats[:,0:64] = h0 ----
  pad_x_kernel<<<(N * 32 + 255) / 256, 256, 0, stream>>>(x, xpad, N);
  gemm(xpad, eW1_t, enc_b1, nullptr, nullptr, 0, henc, 64, N, 64, 32, 1);
  gemm(henc, eW2_t, enc_b2, nullptr, cur, 64, cat, 320, N, 64, 64, 0);

  // ---- transformer layers ----
  for (int l = 0; l < NLAYER; ++l) {
    ln_kernel<<<N / 8, 256, 0, stream>>>(cur, ln1_g + l * 64, ln1_b + l * 64, xn);
    gemm(xn, wq_t + (size_t)l * 512 * 64, nullptr, nullptr, nullptr, 0, qb, 512, N, 512, 64, 0);
    gemm(xn, wk_t + (size_t)l * 512 * 64, nullptr, nullptr, nullptr, 0, kb, 512, N, 512, 64, 0);
    gemm(xn, wv_t + (size_t)l * 512 * 64, nullptr, nullptr, nullptr, 0, vb, 512, N, 512, 64, 0);
    for (int r = 0; r < NROUND; ++r)
      attn_kernel<<<dim3(N / BLK, NHEAD), 256, 0, stream>>>(
          qb, kb, vb, idx + (size_t)r * N, coords, batch, whead + l * 8,
          o3 + (size_t)r * N * HDIM, lse3 + (size_t)r * N * NHEAD);
    combine_kernel<<<(int)(((size_t)N * HDIM) / 256), 256, 0, stream>>>(o3, lse3, att, N);
    gemm(att, wo_t + (size_t)l * 64 * 512, nullptr, cur, cur, 64, nullptr, 0, N, 64, 512, 0);
    ln_kernel<<<N / 8, 256, 0, stream>>>(cur, ln2_g + l * 64, ln2_b + l * 64, xn);
    gemm(xn, ff1_t + (size_t)l * 64 * 64, ff_b1 + l * 64, nullptr, nullptr, 0, ffh, 64, N, 64, 64, 1);
    gemm(ffh, ff2_t + (size_t)l * 64 * 64, ff_b2 + l * 64, cur, cur, 64,
         cat + (l + 1) * 64, 320, N, 64, 64, 0);   // also write feats slice
  }

  // ---- readout ----
  gemm(cat, cat_t, nullptr, nullptr, e2F, 32, e2H, 32, N, 32, 320, 0);
  gemm(e2H, m1_t, mlp_b1, nullptr, nullptr, 0, t1, 32, N, 32, 32, 1);
  gemm(t1, m2_t, mlp_b2, e2F, moF, 32, moH, 32, N, 32, 32, 0);
  gemm(moH, bW1_t, beta_b1, nullptr, nullptr, 0, t2b, 16, N, 16, 32, 1);
  gemm(moH, cW1_t, coord_b1, nullptr, nullptr, 0, t3b, 16, N, 16, 32, 1);
  heads_kernel<<<(N + 255) / 256, 256, 0, stream>>>(t2b, t3b, beta_W2, beta_b2,
                                                    coord_W2, coord_b2,
                                                    (float*)d_out, N);
}
